// NeuralCDE_84507776516480
// MI455X (gfx1250) — matrix operand order repaired and checked by reading, not compiled, not run on hardware
//
#include <hip/hip_runtime.h>
#include <hip/hip_bf16.h>
#include <math.h>

// ---------------------------------------------------------------------------
// Neural CDE (Tsit5, 511 steps) for MI455X / gfx1250.
// Persistent per-batch-tile workgroups (B=256 -> 16 WGs of 16 rows), bf16 WMMA
// with f32 accumulation, all inter-phase sync is workgroup-local.
// 512 threads (16 wave32) per WG. GEMM3 runs transposed (W2 tiles as A,
// h2^T as B) so the D-contraction is an in-lane reduction: 1 shuffle +
// 1 exclusive ds_store per (m,h), no LDS atomics.
// ---------------------------------------------------------------------------

typedef __attribute__((ext_vector_type(16))) __bf16 v16bf;
typedef __attribute__((ext_vector_type(8)))  __bf16 v8bf;
typedef __attribute__((ext_vector_type(8)))  float  v8f;

#define Bsz  256
#define Lts  512
#define Dd   32
#define Hh   128
#define Ww   256
#define LBL  10
#define NSTEP (Lts - 1)
#define NTHR 512

// fragment-packed bf16 weight counts (tiles of 512 bf16 each)
#define W0_TILES (4  * 16)    // K=128 -> 4 kt, N=256 -> 16 nt   (B layout)
#define W1_TILES (8  * 16)    // K=256, N=256                     (B layout)
#define W2_TILES (8  * 256)   // K=256, N=4096                    (A layout)
#define W0_ELEMS (W0_TILES * 512)
#define W1_ELEMS (W1_TILES * 512)
#define W2_ELEMS (W2_TILES * 512)
#define FRAG_ELEMS (W0_ELEMS + W1_ELEMS + W2_ELEMS)

// Tsit5 tableau (fixed step)
__constant__ float c_C[6] = {0.0f, 0.161f, 0.327f, 0.9f, 0.9800255409045097f, 1.0f};
__constant__ float c_A[6][5] = {
  {0.f, 0.f, 0.f, 0.f, 0.f},
  {0.161f, 0.f, 0.f, 0.f, 0.f},
  {-0.008480655492356989f, 0.335480655492357f, 0.f, 0.f, 0.f},
  {2.8971530571054935f, -6.359448489975075f, 4.3622954328695815f, 0.f, 0.f},
  {5.325864828439257f, -11.748883564062828f, 7.4955393428898365f, -0.09249506636175525f, 0.f},
  {5.86145544294642f, -12.92096931784711f, 8.159367898576159f, -0.071584973281401f, -0.028269050394068383f}};
__constant__ float c_B[6] = {0.09646076681806523f, 0.01f, 0.4798896504144996f,
                             1.379008574103742f, -3.290069515436081f, 2.324710524099774f};

// fast epilogue math: v_exp_f32 + v_rcp_f32, no IEEE divide / libm tanh
__device__ __forceinline__ float fast_silu(float x) {
  return x * __builtin_amdgcn_rcpf(1.0f + __expf(-x));
}
__device__ __forceinline__ float fast_tanh(float x) {
  float ax = fabsf(x);
  float e  = __expf(-2.0f * ax);                     // in (0,1], no overflow
  float r  = (1.0f - e) * __builtin_amdgcn_rcpf(1.0f + e);
  return copysignf(r, x);
}

// Activation -> WMMA A-fragment (16-bit A 16x32 ISA layout):
// lanes 0-15 hold K {k..k+7, k+16..k+23}; lanes 16-31 hold {k+8..k+15, k+24..k+31}.
__device__ __forceinline__ v16bf load_afrag(const __bf16* row, int kbase, int hi8) {
  const int b0 = kbase + hi8;
  v8bf lo = *(const v8bf*)(row + b0);
  v8bf hi = *(const v8bf*)(row + b0 + 16);
  v16bf a;
  #pragma unroll
  for (int q = 0; q < 8; ++q) { a[q] = lo[q]; a[q + 8] = hi[q]; }
  return a;
}

// ---------------------------------------------------------------------------
// Pack fp32 weights (row-major (N_out, K_in)) into bf16 WMMA fragments.
//  w0, w1 -> B layout: lane = n_local + 16*h, elem j -> k = 8h+j (j<8), 16+8h+(j-8) (j>=8)
//  w2     -> A layout: lane = n_local + 16*h, elem j -> k = 16h + j (contiguous)
// ---------------------------------------------------------------------------
__global__ void pack_weights(const float* __restrict__ w0,
                             const float* __restrict__ w1,
                             const float* __restrict__ w2,
                             void* __restrict__ dst_raw) {
  __bf16* dst = (__bf16*)dst_raw;
  long id = (long)blockIdx.x * blockDim.x + threadIdx.x;
  if (id >= (long)FRAG_ELEMS) return;
  const float* src; int KT, K; long base; int alayout;
  if (id < (long)W0_ELEMS)                   { src = w0; KT = 4; K = 128; base = 0; alayout = 0; }
  else if (id < (long)(W0_ELEMS + W1_ELEMS)) { src = w1; KT = 8; K = 256; base = W0_ELEMS; id -= W0_ELEMS; alayout = 0; }
  else { src = w2; KT = 8; K = 256; base = W0_ELEMS + W1_ELEMS; id -= (long)W0_ELEMS + W1_ELEMS; alayout = 1; }
  long tile = id >> 9;
  int within = (int)(id & 511);
  int lane = within >> 4;
  int j    = within & 15;
  long nt = tile / KT;
  int  kt = (int)(tile % KT);
  int n = (int)(nt * 16 + (lane & 15));
  int k;
  if (alayout) {
    int koff = (lane >> 4) << 3;                     // 0 or 8
    k = kt * 32 + (koff << 1) + j;                   // contiguous: 2*koff + j
  } else {
    k = kt * 32 + ((lane >> 4) << 3) + ((j < 8) ? j : (8 + j));
  }
  dst[base + (tile << 9) + within] = (__bf16)src[(long)n * K + k];
}

// ---------------------------------------------------------------------------
// Persistent solver kernel: one workgroup (16 wave32) per 16 batch rows.
// ---------------------------------------------------------------------------
__global__ __launch_bounds__(NTHR) void cde_solve(
    const float* __restrict__ ts,
    const float* __restrict__ cdp, const float* __restrict__ ccp, const float* __restrict__ cbp,
    const float* __restrict__ x0,
    const float* __restrict__ l1w, const float* __restrict__ l1b,
    const float* __restrict__ b0v, const float* __restrict__ b1v, const float* __restrict__ b2v,
    const float* __restrict__ l2w, const float* __restrict__ l2b,
    const void* __restrict__ wfrag_raw,
    float* __restrict__ out) {
  extern __shared__ __align__(32) char smem[];
  float* yF    = (float*)smem;          // [16][128]
  float* kS    = yF + 2048;             // [6][16][128]
  float* dXl   = kS + 6 * 2048;         // [16][32]
  float* cbl   = dXl + 512;             // [16][32]
  float* ccl   = cbl + 512;
  float* cdl   = ccl + 512;
  float* logit = cdl + 512;             // [16][10]
  __bf16* aAct = (__bf16*)(logit + 160); // [16][128]  (32B aligned)
  __bf16* h1A  = aAct + 2048;            // [16][256]
  __bf16* h2A  = h1A + 4096;             // [16][256]

  const v16bf* w0t = (const v16bf*)wfrag_raw;
  const v16bf* w1t = w0t + W0_TILES * 32;
  const v16bf* w2t = w1t + W1_TILES * 32;

  const int t    = threadIdx.x;
  const int lane = t & 31;
  const int wv   = t >> 5;              // 0..15
  const int bb0  = blockIdx.x * 16;     // batch-row base for this WG
  const int mRow  = lane & 15;          // fragment row/col index for this lane
  const int khalf = (lane >> 4) << 4;   // B-layout K sub-half (0 or 16)
  const int hi8   = (lane >> 4) << 3;   // A-layout K sub-half (0 or 8)
  const int mbase = (lane >> 4) << 3;   // C-fragment row base (0 or 8)

  // ---- y0 = x0 @ l1_w.T + l1_b ----
  for (int e = t; e < 2048; e += NTHR) {
    int m = e >> 7, hh = e & 127;
    float acc = l1b[hh];
    const float* xr = x0 + (long)(bb0 + m) * Dd;
    const float* wr = l1w + (long)hh * Dd;
    #pragma unroll
    for (int d = 0; d < Dd; ++d) acc += xr[d] * wr[d];
    yF[e] = acc;
  }
  __syncthreads();

  for (int i = 0; i < NSTEP; ++i) {
    const float hdt = ts[i + 1] - ts[i];

    // stream this interval's Hermite coefficients into LDS; prefetch next
    for (int e = t; e < 512; e += NTHR) {
      int m = e >> 5, d = e & 31;
      long gi = (long)(bb0 + m) * (NSTEP * Dd) + (long)i * Dd + d;
      cbl[e] = cbp[gi];
      ccl[e] = ccp[gi];
      cdl[e] = cdp[gi];
      if (i + 1 < NSTEP) {
        __builtin_prefetch(&cbp[gi + Dd], 0, 1);
        __builtin_prefetch(&ccp[gi + Dd], 0, 1);
        __builtin_prefetch(&cdp[gi + Dd], 0, 1);
      }
    }
    __syncthreads();

    for (int s = 0; s < 6; ++s) {
      // ---- phase 1: stage state ytmp -> bf16 A tile; dX ----
      for (int e = t; e < 2048; e += NTHR) {
        float acc = yF[e];
        for (int j = 0; j < s; ++j) acc += hdt * c_A[s][j] * kS[j * 2048 + e];
        aAct[e] = (__bf16)acc;          // e == m*128 + hh (row-major K)
      }
      const float frac = c_C[s] * hdt;
      for (int e = t; e < 512; e += NTHR)
        dXl[e] = cbl[e] + frac * (2.0f * ccl[e] + 3.0f * frac * cdl[e]);
      __syncthreads();

      // ---- GEMM1: [16x128] x [128x256] -> silu -> h1 (bf16); 1 tile/wave ----
      {
        v16bf a[4];
        #pragma unroll
        for (int kt = 0; kt < 4; ++kt)
          a[kt] = load_afrag(aAct + mRow * 128, kt * 32, hi8);
        const int nt = wv;
        v8f c = {};
        #pragma unroll
        for (int kt = 0; kt < 4; ++kt) {
          v16bf b = w0t[(nt * 4 + kt) * 32 + lane];
          c = __builtin_amdgcn_wmma_f32_16x16x32_bf16(false, a[kt], false, b,
                                                      (short)0, c, false, false);
        }
        const int n = nt * 16 + (lane & 15);
        const float bias = b0v[n];
        #pragma unroll
        for (int r = 0; r < 8; ++r)
          h1A[(mbase + r) * 256 + n] = (__bf16)fast_silu(c[r] + bias);
      }
      __syncthreads();

      // ---- GEMM2: [16x256] x [256x256] -> silu -> h2 (bf16); 1 tile/wave ----
      {
        v16bf a[8];
        #pragma unroll
        for (int kt = 0; kt < 8; ++kt)
          a[kt] = load_afrag(h1A + mRow * 256, kt * 32, hi8);
        const int nt = wv;
        v8f c = {};
        #pragma unroll
        for (int kt = 0; kt < 8; ++kt) {
          v16bf b = w1t[(nt * 8 + kt) * 32 + lane];
          c = __builtin_amdgcn_wmma_f32_16x16x32_bf16(false, a[kt], false, b,
                                                      (short)0, c, false, false);
        }
        const int n = nt * 16 + (lane & 15);
        const float bias = b1v[n];
        #pragma unroll
        for (int r = 0; r < 8; ++r)
          h2A[(mbase + r) * 256 + n] = (__bf16)fast_silu(c[r] + bias);
      }
      __syncthreads();

      // ---- GEMM3 (transposed): C = W2_tile[16n x 256k] x h2^T[256k x 16m]
      // C rows = n (h,d), C lanes = batch row m. The D-contraction is an
      // in-lane sum over 8 consecutive d + one shfl_xor(16) per tile pair;
      // each (m,h) is written exactly once -> plain ds_store, no atomics.
      {
        // B = h2^T: lane holds column m = mRow, contiguous K slice (B layout)
        v16bf b[8];
        #pragma unroll
        for (int kt = 0; kt < 8; ++kt)
          b[kt] = *(const v16bf*)&h2A[mRow * 256 + kt * 32 + khalf];
        float* kacc = kS + s * 2048;
        const v8f* dxRow = (const v8f*)&dXl[mRow * 32];   // 4 chunks of 8 d
        for (int tp = 0; tp < 8; ++tp) {
          float ssum = 0.0f;
          #pragma unroll
          for (int half = 0; half < 2; ++half) {
            const int nt = wv * 16 + tp * 2 + half;
            v8f c = {};
            #pragma unroll
            for (int kt = 0; kt < 8; ++kt) {
              v16bf a = w2t[(nt * 8 + kt) * 32 + lane];   // streamed from L2
              c = __builtin_amdgcn_wmma_f32_16x16x32_bf16(false, a, false, b[kt],
                                                          (short)0, c, false, false);
            }
            const v8f bias = *(const v8f*)&b2v[nt * 16 + mbase];
            const v8f dxv  = dxRow[(half << 1) + (mbase >> 3)]; // d = half*16+mbase..+7
            #pragma unroll
            for (int r = 0; r < 8; ++r)
              ssum += fast_tanh(c[r] + bias[r]) * dxv[r];
          }
          ssum += __shfl_xor(ssum, 16, 32);               // join the two n-halves
          const int h = wv * 8 + tp;
          if (lane < 16) kacc[mRow * 128 + h] = ssum;     // exclusive write
        }
      }
      __syncthreads();
    } // stages

    // ---- y update ----
    for (int e = t; e < 2048; e += NTHR) {
      float acc = yF[e];
      #pragma unroll
      for (int j = 0; j < 6; ++j) acc += hdt * c_B[j] * kS[j * 2048 + e];
      yF[e] = acc;
    }
    __syncthreads();
  } // steps

  // ---- readout: softmax(y @ l2_w.T + l2_b) ----
  for (int e = t; e < 16 * LBL; e += NTHR) {
    int m = e / LBL, cc = e % LBL;
    float acc = l2b[cc];
    const float* wr = l2w + (long)cc * Hh;
    #pragma unroll 8
    for (int hh = 0; hh < Hh; ++hh) acc += yF[m * 128 + hh] * wr[hh];
    logit[e] = acc;
  }
  __syncthreads();
  if (t < 16) {
    const int m = t;
    float mx = -1e30f;
    for (int cc = 0; cc < LBL; ++cc) mx = fmaxf(mx, logit[m * LBL + cc]);
    float ex[LBL], sum = 0.0f;
    for (int cc = 0; cc < LBL; ++cc) { ex[cc] = __expf(logit[m * LBL + cc] - mx); sum += ex[cc]; }
    const float inv = __builtin_amdgcn_rcpf(sum);
    for (int cc = 0; cc < LBL; ++cc) out[(long)(bb0 + m) * LBL + cc] = ex[cc] * inv;
  }
}

// ---------------------------------------------------------------------------
extern "C" void kernel_launch(void* const* d_in, const int* in_sizes, int n_in,
                              void* d_out, int out_size, void* d_ws, size_t ws_size,
                              hipStream_t stream) {
  (void)in_sizes; (void)n_in; (void)out_size; (void)ws_size;
  const float* ts  = (const float*)d_in[0];
  const float* cdp = (const float*)d_in[1];
  const float* ccp = (const float*)d_in[2];
  const float* cbp = (const float*)d_in[3];
  // d_in[4] (coeff_a) is unused by the reference vector field
  const float* x0  = (const float*)d_in[5];
  const float* l1w = (const float*)d_in[6];
  const float* l1b = (const float*)d_in[7];
  const float* w0  = (const float*)d_in[8];
  const float* b0  = (const float*)d_in[9];
  const float* w1  = (const float*)d_in[10];
  const float* b1  = (const float*)d_in[11];
  const float* w2  = (const float*)d_in[12];
  const float* b2  = (const float*)d_in[13];
  const float* l2w = (const float*)d_in[14];
  const float* l2b = (const float*)d_in[15];

  // one-shot weight repack (fp32 -> bf16 WMMA fragments) into workspace
  const long total = FRAG_ELEMS;
  pack_weights<<<(int)((total + 255) / 256), 256, 0, stream>>>(w0, w1, w2, d_ws);

  // dynamic LDS: 16544 floats + 10240 bf16 = 86656 bytes (< 320KB WGP budget)
  const size_t smem = 16544 * sizeof(float) + 10240 * sizeof(__bf16);
  cde_solve<<<Bsz / 16, NTHR, smem, stream>>>(ts, cdp, ccp, cbp, x0, l1w, l1b,
                                              b0, b1, b2, l2w, l2b, d_ws,
                                              (float*)d_out);
}
